// Decoder_70592082477367
// MI455X (gfx1250) — compile-verified, matrix operand order checked by reading, not run on hardware
//
#include <hip/hip_runtime.h>
#include <hip/hip_bf16.h>
#include <stddef.h>
#include <stdint.h>

// ---------------------------------------------------------------------------
// Problem constants (match reference)
// ---------------------------------------------------------------------------
#define B_    32
#define TD_   128
#define TE_   128
#define E_    512
#define H_    1024
#define H4_   4096      // 4*H
#define D2E_  1024      // 2*E
#define KOUT_ 2048      // H + 2E
#define TV_   32000

typedef __attribute__((ext_vector_type(16))) __bf16 v16bf;
typedef __attribute__((ext_vector_type(8)))  float  v8f;
typedef __attribute__((ext_vector_type(4)))  unsigned int v4u;
typedef __attribute__((ext_vector_type(8)))  int v8i;
typedef __attribute__((ext_vector_type(4)))  int v4i;

__device__ __forceinline__ float sigmoidf_(float x) { return 1.0f / (1.0f + __expf(-x)); }

// ---------------------------------------------------------------------------
// f32 -> bf16 conversion (grid-stride)
// ---------------------------------------------------------------------------
__global__ void k_f32_to_bf16(const float* __restrict__ src, __bf16* __restrict__ dst, int n) {
    int i = blockIdx.x * blockDim.x + threadIdx.x;
    int stride = gridDim.x * blockDim.x;
    for (; i < n; i += stride) dst[i] = (__bf16)src[i];
}

// ---------------------------------------------------------------------------
// Encoder embeddings: enc[b,e,d] = concat(enc_emb[src], pos_emb[pos]); also
// write the transpose encT[b,d,e] (used as the N-major operand of attn@V).
// ---------------------------------------------------------------------------
__global__ void k_embed_enc(const int* __restrict__ src, const int* __restrict__ pos,
                            const float* __restrict__ enc_emb, const float* __restrict__ pos_emb,
                            __bf16* __restrict__ enc, __bf16* __restrict__ encT) {
    int be = blockIdx.x;                 // b*TE + e
    int b = be / TE_, e = be % TE_;
    int sid = src[be];
    int pid = pos[be];
    for (int d = threadIdx.x; d < D2E_; d += blockDim.x) {
        float v = (d < E_) ? enc_emb[(size_t)sid * E_ + d] : pos_emb[(size_t)pid * E_ + (d - E_)];
        __bf16 bv = (__bf16)v;
        enc [((size_t)b * TE_ + e) * D2E_ + d] = bv;
        encT[((size_t)b * D2E_ + d) * TE_ + e] = bv;
    }
}

// Decoder word embeddings -> words_bf16 (B*TD, E)
__global__ void k_embed_dec(const int* __restrict__ tgt, const float* __restrict__ dec_emb,
                            __bf16* __restrict__ words) {
    int m = blockIdx.x;                  // b*TD + t
    int id = tgt[m];
    for (int d = threadIdx.x; d < E_; d += blockDim.x)
        words[(size_t)m * E_ + d] = (__bf16)dec_emb[(size_t)id * E_ + d];
}

// avg[b,d] = mean_e enc[b,e,d]
__global__ void k_avg(const __bf16* __restrict__ enc, float* __restrict__ avg) {
    int i = blockIdx.x * blockDim.x + threadIdx.x;   // b*D2E + d
    if (i >= B_ * D2E_) return;
    int b = i / D2E_, d = i % D2E_;
    float s = 0.f;
    for (int e = 0; e < TE_; ++e) s += (float)enc[((size_t)b * TE_ + e) * D2E_ + d];
    avg[i] = s * (1.0f / TE_);
}

// hidden = avg @ W_h0.T + b_h0 ; seeds c0 (f32), h0 (bf16), and q[:,0,:]
__global__ void k_hidden(const float* __restrict__ avg, const float* __restrict__ W_h0,
                         const float* __restrict__ b_h0,
                         float* __restrict__ c, __bf16* __restrict__ hbf, __bf16* __restrict__ q) {
    int i = blockIdx.x * blockDim.x + threadIdx.x;   // b*H + h
    if (i >= B_ * H_) return;
    int b = i / H_, h = i % H_;
    const float* a = avg + (size_t)b * D2E_;
    const float* w = W_h0 + (size_t)h * D2E_;
    float s = b_h0[h];
    for (int k = 0; k < D2E_; ++k) s += a[k] * w[k];
    c[i] = s;
    __bf16 hb = (__bf16)s;
    hbf[i] = hb;
    q[((size_t)b * TD_) * H_ + h] = hb;
}

// ---------------------------------------------------------------------------
// Generic batched bf16 WMMA GEMM:  C(MxN,f32) = A(MxK) * B(NxK)^T [+bias0+bias1][+Cadd]
// Block 256 threads = 8 waves (2x4), block tile 64(M)x128(N), wave tile 32x32
// (2x2 fragments of v_wmma_f32_16x16x32_bf16). Software-pipelined k-loop with
// the FINAL iteration peeled so the next-fragment loads and register rotation
// are unconditional (no cndmask predication in the hot loop). Requires
// K >= 64 and K % 32 == 0 (true for every call site: 512/1024/2048/128).
// Loads are index-clamped so EXEC stays all-ones (WMMA requirement).
// ---------------------------------------------------------------------------
__global__ void k_gemm_bf16(const __bf16* __restrict__ A, int lda, int bsA,
                            const __bf16* __restrict__ Bm, int ldb, int bsB,
                            float* __restrict__ C, int ldc, int bsC,
                            const float* __restrict__ Cadd, int ldcadd, int bsCadd,
                            const float* __restrict__ bias0, const float* __restrict__ bias1,
                            int M, int N, int K) {
    int batch = blockIdx.z;
    A  += (size_t)batch * bsA;
    Bm += (size_t)batch * bsB;
    C  += (size_t)batch * bsC;
    if (Cadd) Cadd += (size_t)batch * bsCadd;

    int lane = threadIdx.x & 31;
    int wave = threadIdx.x >> 5;
    int wm = wave & 1;                   // 2 waves in M
    int wn = wave >> 1;                  // 4 waves in N
    int m0 = blockIdx.y * 64 + wm * 32;
    int n0 = blockIdx.x * 128 + wn * 32;

    int r    = lane & 15;
    int koff = (lane >> 4) * 16;

    const __bf16* ap[2];
    const __bf16* bp[2];
    for (int i = 0; i < 2; ++i) {
        int m = m0 + i * 16 + r; m = (m < M) ? m : (M - 1);
        int n = n0 + i * 16 + r; n = (n < N) ? n : (N - 1);
        ap[i] = A  + (size_t)m * lda + koff;
        bp[i] = Bm + (size_t)n * ldb + koff;
    }

    v8f zero = {};
    v8f acc[2][2];
    for (int i = 0; i < 2; ++i)
        for (int j = 0; j < 2; ++j) acc[i][j] = zero;

    v16bf a[2], b[2], an[2], bn[2];
    for (int i = 0; i < 2; ++i) { a[i] = *(const v16bf*)(ap[i]); b[i] = *(const v16bf*)(bp[i]); }

    // pipelined main loop: iterations [0, K-32); final iteration peeled below
    for (int k0 = 0; k0 < K - 32; k0 += 32) {
        for (int i = 0; i < 2; ++i) {
            an[i] = *(const v16bf*)(ap[i] + k0 + 32);
            bn[i] = *(const v16bf*)(bp[i] + k0 + 32);
        }
        __builtin_prefetch((const void*)(ap[0] + k0 + 64), 0, 1);
        __builtin_prefetch((const void*)(bp[0] + k0 + 64), 0, 1);
        for (int i = 0; i < 2; ++i)
            for (int j = 0; j < 2; ++j)
                acc[i][j] = __builtin_amdgcn_wmma_f32_16x16x32_bf16(
                    false, a[i], false, b[j], (short)0, acc[i][j], false, false);
        for (int i = 0; i < 2; ++i) { a[i] = an[i]; b[i] = bn[i]; }
    }
    // peeled final k-step
    for (int i = 0; i < 2; ++i)
        for (int j = 0; j < 2; ++j)
            acc[i][j] = __builtin_amdgcn_wmma_f32_16x16x32_bf16(
                false, a[i], false, b[j], (short)0, acc[i][j], false, false);

    int mhi = (lane >> 4) * 8;
    int nl  = lane & 15;
    for (int j = 0; j < 2; ++j) {
        int n = n0 + j * 16 + nl;
        if (n >= N) continue;
        float bn2 = 0.f;
        if (bias0) bn2 += bias0[n];
        if (bias1) bn2 += bias1[n];
        for (int i = 0; i < 2; ++i) {
            int mbase = m0 + i * 16 + mhi;
            for (int v = 0; v < 8; ++v) {
                int m = mbase + v;
                if (m < M) {
                    float val = acc[i][j][v] + bn2;
                    if (Cadd) val += Cadd[(size_t)m * ldcadd + n];
                    C[(size_t)m * ldc + n] = val;
                }
            }
        }
    }
}

// ---------------------------------------------------------------------------
// Tensor Data Mover helper: DMA a (tile_rows x 32) bf16 2-D tile (row stride
// row_stride_elems) from global into LDS at lds_off. D# packing per CDNA5 ISA
// 8.3/8.4: group0 = {count=1, lds_addr, global_addr[56:0], type=2};
// group1 = {data_size=2B, tensor_dim0=k_len, tensor_dim1=rows, tile_dim0=32,
// tile_dim1=tile_rows, dim0_stride}. Per-wave op, EXEC ignored -> call from
// one wave only. Tracked with TENSORcnt. 6-arg builtin on this toolchain:
// (v4u, v8i, v4i, v4i, v8i, i32 cpol) -- trailing vector zeroed (2-D tile).
// ---------------------------------------------------------------------------
__device__ __forceinline__ void tdm_load_tile(const __bf16* gsrc, unsigned lds_off,
                                              int k_len, int tensor_rows,
                                              int row_stride_elems, int tile_rows) {
    unsigned long long ga = (unsigned long long)(uintptr_t)gsrc;
    v4u g0;
    g0[0] = 1u;                                          // count=1, user descriptor
    g0[1] = lds_off;                                     // lds_addr (bytes)
    g0[2] = (unsigned)(ga & 0xffffffffu);                // global_addr[31:0]
    g0[3] = (unsigned)((ga >> 32) & 0x01ffffffu) | (2u << 30);  // addr[56:32], type=2
    v8i g1;
    g1[0] = (int)(1u << 16);                             // workgroup_mask=0, data_size=2B
    g1[1] = (int)(((unsigned)k_len & 0xffffu) << 16);    // tensor_dim0[15:0]
    g1[2] = (int)((((unsigned)k_len >> 16) & 0xffffu) |
                  (((unsigned)tensor_rows & 0xffffu) << 16));   // dim0 hi | dim1 lo
    g1[3] = (int)((((unsigned)tensor_rows >> 16) & 0xffffu) |
                  (32u << 16));                          // dim1 hi | tile_dim0=32
    g1[4] = (int)((unsigned)tile_rows & 0xffffu);        // tile_dim1 | tile_dim2=0
    g1[5] = (int)(unsigned)row_stride_elems;             // tensor_dim0_stride[31:0]
    g1[6] = 0;                                           // stride0 hi | stride1 lo
    g1[7] = 0;
    v4i g2 = {0, 0, 0, 0};
    v4i g3 = {0, 0, 0, 0};
    v8i g4 = {0, 0, 0, 0, 0, 0, 0, 0};
    __builtin_amdgcn_tensor_load_to_lds(g0, g1, g2, g3, g4, 0);
}

__device__ __forceinline__ unsigned lds_offset_of(const void* p) {
    // LDS flat aperture: addr[31:0] is the in-LDS byte offset.
    return (unsigned)(uintptr_t)p;
}

// ---------------------------------------------------------------------------
// Dedicated output-projection GEMM (M=4096, N=32000, K=2048): TDM stages
// double-buffered A (64x32) and B (128x32) bf16 tiles into LDS; all 8 waves
// compute 2x2 WMMA fragments from LDS. Exact tiling -> EXEC all-ones, no
// clamps. Sync: wave0 issues tensor_load_to_lds, s_wait_tensorcnt 0, then
// workgroup barrier publishes the buffer.
// ---------------------------------------------------------------------------
#define OBM 64
#define OBN 128
#define OBK 32

__global__ void k_gemm_out_tdm(const __bf16* __restrict__ A,   // (4096, 2048)
                               const __bf16* __restrict__ Bm,  // (32000, 2048)
                               const float* __restrict__ bias, // (32000)
                               float* __restrict__ C) {        // (4096, 32000)
    __shared__ __align__(64) __bf16 smA[2][OBM * OBK];   // 4 KB x2
    __shared__ __align__(64) __bf16 smB[2][OBN * OBK];   // 8 KB x2

    const int M = B_ * TD_, N = TV_, K = KOUT_;
    int lane = threadIdx.x & 31;
    int wave = threadIdx.x >> 5;
    int wm = wave & 1;
    int wn = wave >> 1;
    int m0 = blockIdx.y * OBM;
    int n0 = blockIdx.x * OBN;

    int r    = lane & 15;
    int koff = (lane >> 4) * 16;

    // LDS fragment pointers (per wave, per m/n sub-tile)
    const __bf16* la[2][2];   // [buf][i]
    const __bf16* lb[2][2];
    for (int bu = 0; bu < 2; ++bu)
        for (int i = 0; i < 2; ++i) {
            la[bu][i] = &smA[bu][(wm * 32 + i * 16 + r) * OBK + koff];
            lb[bu][i] = &smB[bu][(wn * 32 + i * 16 + r) * OBK + koff];
        }

    v8f zero = {};
    v8f acc[2][2];
    for (int i = 0; i < 2; ++i)
        for (int j = 0; j < 2; ++j) acc[i][j] = zero;

    // prologue: DMA first k-tile into buffer 0
    if (wave == 0) {
        tdm_load_tile(A  + (size_t)m0 * K, lds_offset_of(&smA[0][0]), K, M, K, OBM);
        tdm_load_tile(Bm + (size_t)n0 * K, lds_offset_of(&smB[0][0]), K, N, K, OBN);
    }

    int buf = 0;
    for (int k0 = 0; k0 < K; k0 += OBK) {
        if (wave == 0) __builtin_amdgcn_s_wait_tensorcnt(0);   // current buffer landed
        __syncthreads();                                       // publish; prev buf free
        if (wave == 0 && (k0 + OBK) < K) {
            tdm_load_tile(A  + (size_t)m0 * K + (k0 + OBK), lds_offset_of(&smA[buf ^ 1][0]),
                          K, M, K, OBM);
            tdm_load_tile(Bm + (size_t)n0 * K + (k0 + OBK), lds_offset_of(&smB[buf ^ 1][0]),
                          K, N, K, OBN);
        }
        v16bf a[2], b[2];
        for (int i = 0; i < 2; ++i) {
            a[i] = *(const v16bf*)(la[buf][i]);
            b[i] = *(const v16bf*)(lb[buf][i]);
        }
        for (int i = 0; i < 2; ++i)
            for (int j = 0; j < 2; ++j)
                acc[i][j] = __builtin_amdgcn_wmma_f32_16x16x32_bf16(
                    false, a[i], false, b[j], (short)0, acc[i][j], false, false);
        __syncthreads();                                       // readers done before reuse
        buf ^= 1;
    }

    int mhi = (lane >> 4) * 8;
    int nl  = lane & 15;
    for (int j = 0; j < 2; ++j) {
        int n = n0 + wn * 32 + j * 16 + nl;
        float bn = bias[n];
        for (int i = 0; i < 2; ++i) {
            int mbase = m0 + wm * 32 + i * 16 + mhi;
            for (int v = 0; v < 8; ++v)
                C[(size_t)(mbase + v) * N + n] = acc[i][j][v] + bn;
        }
    }
}

// ---------------------------------------------------------------------------
// LSTM pointwise gates for one timestep (torch gate order i,f,g,o)
// ---------------------------------------------------------------------------
__global__ void k_lstm_point(const float* __restrict__ z, float* __restrict__ c,
                             __bf16* __restrict__ hbf, __bf16* __restrict__ q,
                             __bf16* __restrict__ combined, int t) {
    int i = blockIdx.x * blockDim.x + threadIdx.x;   // b*H + j
    if (i >= B_ * H_) return;
    int b = i / H_, j = i % H_;
    const float* zb = z + (size_t)b * H4_;
    float zi = zb[j], zf = zb[j + H_], zg = zb[j + 2 * H_], zo = zb[j + 3 * H_];
    float cc = sigmoidf_(zf) * c[i] + sigmoidf_(zi) * tanhf(zg);
    float h  = sigmoidf_(zo) * tanhf(cc);
    c[i] = cc;
    __bf16 hb = (__bf16)h;
    hbf[i] = hb;
    combined[((size_t)b * TD_ + t) * KOUT_ + j] = hb;          // lstm_out half
    if (t + 1 < TD_) q[((size_t)b * TD_ + (t + 1)) * H_ + j] = hb;
}

// ---------------------------------------------------------------------------
// softmax -> mask -> renormalize  ==  masked softmax (one wave32 per row)
// ---------------------------------------------------------------------------
__global__ void k_softmax(const float* __restrict__ energy, const int* __restrict__ slen,
                          __bf16* __restrict__ attn) {
    int lane = threadIdx.x & 31;
    int wave = threadIdx.x >> 5;
    int row = blockIdx.x * 8 + wave;     // b*TD + d
    int b = row / TD_;
    int sl = slen[b];
    const float* e = energy + (size_t)row * TE_;
    float v[4]; bool msk[4];
    float mx = -3.0e38f;
    for (int qq = 0; qq < 4; ++qq) {
        int idx = lane + qq * 32;
        bool ok = idx < sl;
        float x = ok ? e[idx] : -3.0e38f;
        v[qq] = x; msk[qq] = ok;
        mx = fmaxf(mx, x);
    }
    for (int off = 16; off > 0; off >>= 1) mx = fmaxf(mx, __shfl_xor(mx, off, 32));
    float s = 0.f;
    for (int qq = 0; qq < 4; ++qq) {
        float ex = msk[qq] ? __expf(v[qq] - mx) : 0.f;
        v[qq] = ex; s += ex;
    }
    for (int off = 16; off > 0; off >>= 1) s += __shfl_xor(s, off, 32);
    float inv = 1.0f / s;
    __bf16* o = attn + (size_t)row * TE_;
    for (int qq = 0; qq < 4; ++qq) o[lane + qq * 32] = (__bf16)(v[qq] * inv);
}

// context f32 -> combined[:, :, H:H+2E] bf16
__global__ void k_cvt_ctx(const float* __restrict__ ctx, __bf16* __restrict__ combined) {
    int n = B_ * TD_ * D2E_;
    int stride = gridDim.x * blockDim.x;
    for (int i = blockIdx.x * blockDim.x + threadIdx.x; i < n; i += stride) {
        int row = i / D2E_, d = i % D2E_;
        combined[(size_t)row * KOUT_ + H_ + d] = (__bf16)ctx[i];
    }
}

// ---------------------------------------------------------------------------
// Host orchestration
// ---------------------------------------------------------------------------
extern "C" void kernel_launch(void* const* d_in, const int* in_sizes, int n_in,
                              void* d_out, int out_size, void* d_ws, size_t ws_size,
                              hipStream_t stream) {
    const int*   tgt     = (const int*)d_in[0];
    const int*   slen    = (const int*)d_in[2];
    const int*   ssrc    = (const int*)d_in[3];
    const int*   pos     = (const int*)d_in[4];
    const float* enc_emb = (const float*)d_in[6];
    const float* pos_emb = (const float*)d_in[7];
    const float* dec_emb = (const float*)d_in[8];
    const float* W_ih    = (const float*)d_in[9];
    const float* W_hh    = (const float*)d_in[10];
    const float* b_ih    = (const float*)d_in[11];
    const float* b_hh    = (const float*)d_in[12];
    const float* W_h0    = (const float*)d_in[13];
    const float* b_h0    = (const float*)d_in[14];
    const float* W_out   = (const float*)d_in[15];
    const float* b_out   = (const float*)d_in[16];
    float* out = (float*)d_out;

    // ---- workspace carve (256B aligned sub-buffers) ----
    char* wp = (char*)d_ws;
    auto carve = [&](size_t bytes) -> char* {
        char* p = wp; wp += (bytes + 255) & ~(size_t)255; return p;
    };
    __bf16* enc_bf  = (__bf16*)carve((size_t)B_ * TE_ * D2E_ * 2);   // (B,TE,2E)
    __bf16* encT_bf = (__bf16*)carve((size_t)B_ * D2E_ * TE_ * 2);   // (B,2E,TE)
    __bf16* words   = (__bf16*)carve((size_t)B_ * TD_ * E_ * 2);     // (B*TD,E)
    __bf16* Wih_bf  = (__bf16*)carve((size_t)H4_ * E_ * 2);
    __bf16* Whh_bf  = (__bf16*)carve((size_t)H4_ * H_ * 2);
    __bf16* Wout_bf = (__bf16*)carve((size_t)TV_ * KOUT_ * 2);
    __bf16* q_bf    = (__bf16*)carve((size_t)B_ * TD_ * H_ * 2);     // lstm_to_attn
    __bf16* h_bf    = (__bf16*)carve((size_t)B_ * H_ * 2);
    __bf16* attn_bf = (__bf16*)carve((size_t)B_ * TD_ * TE_ * 2);
    __bf16* comb_bf = (__bf16*)carve((size_t)B_ * TD_ * KOUT_ * 2);  // [lstm_out|ctx]
    float*  avg     = (float*)carve((size_t)B_ * D2E_ * 4);
    float*  c_f     = (float*)carve((size_t)B_ * H_ * 4);
    float*  z_f     = (float*)carve((size_t)B_ * H4_ * 4);
    float*  xpre    = (float*)carve((size_t)B_ * TD_ * H4_ * 4);     // (B,TD,4H)
    float*  energy  = (float*)carve((size_t)B_ * TD_ * TE_ * 4);
    float*  ctx     = (float*)carve((size_t)B_ * TD_ * D2E_ * 4);

    // ---- weight conversions to bf16 (grid-stride) ----
    k_f32_to_bf16<<<2048, 256, 0, stream>>>(W_ih,  Wih_bf,  H4_ * E_);
    k_f32_to_bf16<<<2048, 256, 0, stream>>>(W_hh,  Whh_bf,  H4_ * H_);
    k_f32_to_bf16<<<8192, 256, 0, stream>>>(W_out, Wout_bf, TV_ * KOUT_);

    // ---- embeddings ----
    k_embed_enc<<<B_ * TE_, 256, 0, stream>>>(ssrc, pos, enc_emb, pos_emb, enc_bf, encT_bf);
    k_embed_dec<<<B_ * TD_, 256, 0, stream>>>(tgt, dec_emb, words);

    // ---- initial hidden ----
    k_avg<<<(B_ * D2E_ + 255) / 256, 256, 0, stream>>>(enc_bf, avg);
    k_hidden<<<(B_ * H_ + 255) / 256, 256, 0, stream>>>(avg, W_h0, b_h0, c_f, h_bf, q_bf);

    // ---- x_pre = words @ W_ih.T + b_ih + b_hh   (M=4096, N=4096, K=512) ----
    {
        dim3 g(H4_ / 128, (B_ * TD_ + 63) / 64, 1);
        k_gemm_bf16<<<g, 256, 0, stream>>>(words, E_, 0, Wih_bf, E_, 0,
                                           xpre, H4_, 0,
                                           nullptr, 0, 0, b_ih, b_hh,
                                           B_ * TD_, H4_, E_);
    }

    // ---- LSTM recurrence: 128 sequential (GEMM + pointwise) steps ----
    for (int t = 0; t < TD_; ++t) {
        dim3 g(H4_ / 128, 1, 1);   // M=32 fits one 64-row block (clamped loads)
        k_gemm_bf16<<<g, 256, 0, stream>>>(h_bf, H_, 0, Whh_bf, H_, 0,
                                           z_f, H4_, 0,
                                           xpre + (size_t)t * H4_, TD_ * H4_, 0,
                                           nullptr, nullptr,
                                           B_, H4_, H_);
        k_lstm_point<<<(B_ * H_ + 255) / 256, 256, 0, stream>>>(z_f, c_f, h_bf, q_bf, comb_bf, t);
    }

    // ---- attention energy: per-batch  Q(TD,H) @ enc(TE,H)^T ----
    {
        dim3 g(TE_ / 128, TD_ / 64, B_);
        k_gemm_bf16<<<g, 256, 0, stream>>>(q_bf, H_, TD_ * H_, enc_bf, D2E_, TE_ * D2E_,
                                           energy, TE_, TD_ * TE_,
                                           nullptr, 0, 0, nullptr, nullptr,
                                           TD_, TE_, H_);
    }

    // ---- masked softmax ----
    k_softmax<<<(B_ * TD_) / 8, 256, 0, stream>>>(energy, slen, attn_bf);

    // ---- context: per-batch  attn(TD,TE) @ encT(2E,TE)^T ----
    {
        dim3 g(D2E_ / 128, TD_ / 64, B_);
        k_gemm_bf16<<<g, 256, 0, stream>>>(attn_bf, TE_, TD_ * TE_, encT_bf, TE_, D2E_ * TE_,
                                           ctx, D2E_, TD_ * D2E_,
                                           nullptr, 0, 0, nullptr, nullptr,
                                           TD_, D2E_, TE_);
    }
    k_cvt_ctx<<<4096, 256, 0, stream>>>(ctx, comb_bf);

    // ---- output projection via TDM + LDS double buffering ----
    {
        dim3 g(TV_ / OBN, (B_ * TD_) / OBM, 1);   // (250, 64)
        k_gemm_out_tdm<<<g, 256, 0, stream>>>(comb_bf, Wout_bf, b_out, out);
    }

    (void)in_sizes; (void)n_in; (void)out_size; (void)ws_size;
}
